// Block_16690242913196
// MI455X (gfx1250) — compile-verified, matrix operand order checked by reading, not run on hardware
//
#include <hip/hip_runtime.h>

// ---------------------------------------------------------------------------
// Types
// ---------------------------------------------------------------------------
typedef unsigned short u16;
typedef unsigned int   u32;
typedef __attribute__((ext_vector_type(4)))  u32    u32x4;
typedef __attribute__((ext_vector_type(8)))  float  f32x8;
typedef __attribute__((ext_vector_type(16))) __bf16 bf16x16;

#define LDS_STRIDE 40   // halfs: multiple of 8 (16B align), 20 dwords -> conflict-free b128 reads

__device__ __forceinline__ u16 f2bf(float f) {
    union { float f; u32 u; } c; c.f = f;
    u32 r = c.u + 0x7FFFu + ((c.u >> 16) & 1u);   // round-to-nearest-even
    return (u16)(r >> 16);
}

__device__ __forceinline__ f32x8 zero8() {
    f32x8 z = {0.f, 0.f, 0.f, 0.f, 0.f, 0.f, 0.f, 0.f};
    return z;
}

// Low 32 bits of a generic pointer into LDS = wave-relative LDS byte offset
// (per the ISA aperture rule).
__device__ __forceinline__ u32 lds_off(const void* p) {
    return (u32)(uintptr_t)p;
}

// CDNA5 async global->LDS copy (ASYNCcnt-tracked, no VGPR transit).
__device__ __forceinline__ void async_load_b128(u32 lds, const u16* g) {
    asm volatile("global_load_async_to_lds_b128 %0, %1, off"
                 :: "v"(lds), "v"(g) : "memory");
}
__device__ __forceinline__ void wait_async() {
    asm volatile("s_wait_asynccnt 0x0" ::: "memory");
}

// Load one 16x16x32 bf16 WMMA A/B fragment. Caller passes the address of
// half index (row*stride + khalf [+kk]); second chunk is +16 halfs per the
// ISA 16-bit A-matrix layout (lanes 0-15: K 0-7 & 16-23; lanes 16-31: +8).
__device__ __forceinline__ bf16x16 ld_frag(const u16* p) {
    union { bf16x16 v; u32x4 q[2]; } f;
    f.q[0] = *(const u32x4*)(p);
    f.q[1] = *(const u32x4*)(p + 16);
    return f.v;
}

__device__ __forceinline__ f32x8 wmma_bf16(bf16x16 a, bf16x16 b, f32x8 c) {
    return __builtin_amdgcn_wmma_f32_16x16x32_bf16(
        /*neg_a=*/false, a, /*neg_b=*/false, b,
        /*c_mod=*/(short)0, c, /*reuse_a=*/false, /*reuse_b=*/false);
}

__device__ __forceinline__ float gelu_new(float v) {
    float x3 = v * v * v;
    return 0.5f * v * (1.0f + tanhf(0.7978845608028654f * (v + 0.044715f * x3)));
}

// ---------------------------------------------------------------------------
// f32 -> bf16 weight conversion
// ---------------------------------------------------------------------------
__global__ void convert_kernel(const float* __restrict__ in, u16* __restrict__ out, int n) {
    for (int i = blockIdx.x * blockDim.x + threadIdx.x; i < n; i += gridDim.x * blockDim.x)
        out[i] = f2bf(in[i]);
}

// ---------------------------------------------------------------------------
// LayerNorm: f32 [rows, D] -> bf16 [rows, D]
// ---------------------------------------------------------------------------
__global__ void ln_kernel(const float* __restrict__ x, const float* __restrict__ g,
                          const float* __restrict__ bb, u16* __restrict__ o, int D) {
    int row = blockIdx.x;
    const float* xr = x + (size_t)row * D;
    float s = 0.f, s2 = 0.f;
    for (int i = threadIdx.x; i < D; i += blockDim.x) { float v = xr[i]; s += v; s2 += v * v; }
#pragma unroll
    for (int m = 16; m >= 1; m >>= 1) { s += __shfl_xor(s, m, 32); s2 += __shfl_xor(s2, m, 32); }
    __shared__ float a1[8], a2[8];
    int w = threadIdx.x >> 5;
    if ((threadIdx.x & 31) == 0) { a1[w] = s; a2[w] = s2; }
    __syncthreads();
    float ts = 0.f, ts2 = 0.f;
#pragma unroll
    for (int i = 0; i < 8; i++) { ts += a1[i]; ts2 += a2[i]; }
    float mu   = ts / (float)D;
    float var  = ts2 / (float)D - mu * mu;
    float rstd = rsqrtf(var + 1e-5f);
    u16* orow = o + (size_t)row * D;
    for (int i = threadIdx.x; i < D; i += blockDim.x)
        orow[i] = f2bf((xr[i] - mu) * rstd * g[i] + bb[i]);
}

// ---------------------------------------------------------------------------
// WMMA GEMM: C[M,N] = A[M,K](bf16) @ W[K,N](bf16) + bias [+ epilogue]
//   epi 0: store bf16
//   epi 1: GELU then store bf16
//   epi 2: add f32 residual, store f32
// Tile 128x128, BK=32, 256 threads = 8 waves, each wave 32x64 (2x4 frags).
// Double-buffered LDS; A tile staged via async global->LDS (ASYNCcnt),
// W tile transposed through VGPRs into LDS.
// ---------------------------------------------------------------------------
__launch_bounds__(256, 2)
__global__ void gemm_bf16(const u16* __restrict__ A, const u16* __restrict__ W,
                          const float* __restrict__ bias, const float* __restrict__ res,
                          float* __restrict__ outF, u16* __restrict__ outB,
                          int M, int N, int K, int epi) {
    __shared__ __align__(16) u16 sA[2][128 * LDS_STRIDE];
    __shared__ __align__(16) u16 sB[2][128 * LDS_STRIDE];   // holds W^T tile: [n][k]

    const int tid   = threadIdx.x;
    const int lane  = tid & 31;
    const int wave  = tid >> 5;
    const int wm    = wave & 3;             // 4 row groups of 32
    const int wn    = wave >> 2;            // 2 col groups of 64
    const int lm    = lane & 15;
    const int khalf = (lane >> 4) << 3;     // 0 or 8

    const int rowBase = blockIdx.y * 128;
    const int colBase = blockIdx.x * 128;

    // global->LDS load mapping
    const int ar  = tid >> 1;               // A row 0..127
    const int ac  = (tid & 1) << 4;         // 0 or 16
    const int wk  = tid & 31;               // W k-row 0..31
    const int wn0 = (tid >> 5) << 4;        // W col group * 16

    const u16* gA = A + (size_t)(rowBase + ar) * K + ac;   // + k0 per tile

    f32x8 acc[2][4];
#pragma unroll
    for (int i = 0; i < 2; i++)
#pragma unroll
        for (int j = 0; j < 4; j++) acc[i][j] = zero8();

    // ---- prologue: stage tile 0 into buffer 0 ----
    {
        async_load_b128(lds_off(&sA[0][ar * LDS_STRIDE + ac]),     gA);
        async_load_b128(lds_off(&sA[0][ar * LDS_STRIDE + ac + 8]), gA + 8);
        const u16* g = W + (size_t)wk * N + colBase + wn0;
        union { u32x4 q[2]; u16 h[16]; } t;
        t.q[0] = *(const u32x4*)(g);
        t.q[1] = *(const u32x4*)(g + 8);
#pragma unroll
        for (int j = 0; j < 16; j++)
            sB[0][(wn0 + j) * LDS_STRIDE + wk] = t.h[j];
    }

    int buf = 0;
    for (int k0 = 0; k0 < K; k0 += 32) {
        wait_async();           // own async writes to LDS complete
        __syncthreads();        // publish buffer `buf` (also drains ds stores)

        // Fragment loads for the CURRENT buffer first, so the s_wait_dscnt
        // guarding the WMMAs does not cover the next tile's ds stores.
        bf16x16 af[2], bfr[4];
#pragma unroll
        for (int i = 0; i < 2; i++)
            af[i] = ld_frag(&sA[buf][(wm * 32 + i * 16 + lm) * LDS_STRIDE + khalf]);
#pragma unroll
        for (int j = 0; j < 4; j++)
            bfr[j] = ld_frag(&sB[buf][(wn * 64 + j * 16 + lm) * LDS_STRIDE + khalf]);

        // Stage the NEXT tile into the other buffer (overlaps with WMMAs).
        if (k0 + 32 < K) {
            const u16* ga = gA + k0 + 32;
            async_load_b128(lds_off(&sA[buf ^ 1][ar * LDS_STRIDE + ac]),     ga);
            async_load_b128(lds_off(&sA[buf ^ 1][ar * LDS_STRIDE + ac + 8]), ga + 8);
            const u16* g = W + (size_t)(k0 + 32 + wk) * N + colBase + wn0;
            union { u32x4 q[2]; u16 h[16]; } t;
            t.q[0] = *(const u32x4*)(g);
            t.q[1] = *(const u32x4*)(g + 8);
#pragma unroll
            for (int j = 0; j < 16; j++)
                sB[buf ^ 1][(wn0 + j) * LDS_STRIDE + wk] = t.h[j];
        }

#pragma unroll
        for (int i = 0; i < 2; i++)
#pragma unroll
            for (int j = 0; j < 4; j++)
                acc[i][j] = wmma_bf16(af[i], bfr[j], acc[i][j]);
        buf ^= 1;
    }

    // Epilogue. C layout: lane<16 -> M=e, lane>=16 -> M=8+e; N = lm.
    const int mh = (lane >> 4) << 3;
#pragma unroll
    for (int i = 0; i < 2; i++) {
#pragma unroll
        for (int j = 0; j < 4; j++) {
            int col  = colBase + wn * 64 + j * 16 + lm;
            float bv = bias[col];
#pragma unroll
            for (int e = 0; e < 8; e++) {
                int row = rowBase + wm * 32 + i * 16 + mh + e;
                float v = acc[i][j][e] + bv;
                if (epi == 1) v = gelu_new(v);
                size_t off = (size_t)row * N + col;
                if (epi == 2) outF[off] = v + res[off];
                else          outB[off] = f2bf(v);
            }
        }
    }
}

// ---------------------------------------------------------------------------
// Causal flash attention.
// qkv: bf16 [B*S, 3*D] rows = (b*S+s), cols = {q|k|v}*D + h*64 + dh
// aout: bf16 [B*S, D]
// Grid: (S/64, B*H); block 128 = 4 waves, each wave owns 16 query rows.
// Q and K tiles staged via async global->LDS; V transposed through VGPRs.
// ---------------------------------------------------------------------------
__launch_bounds__(128, 2)
__global__ void attn_kernel(const u16* __restrict__ qkv, u16* __restrict__ aout) {
    const int S = 2048, D = 1024, DH = 64, TD = 3072;

    __shared__ __align__(16) u16 qs[64 * 72];            // Q tile  [64 q][64 dh]
    __shared__ __align__(16) u16 ks[32 * 72];            // K tile  [32 key][64 dh]
    __shared__ __align__(16) u16 vts[64 * LDS_STRIDE];   // V^T     [64 dh][32 key]
    __shared__ __align__(16) u16 ps[64 * LDS_STRIDE];    // P per wave: [wave*16+m][32 key]

    const int tid   = threadIdx.x;
    const int lane  = tid & 31;
    const int wave  = tid >> 5;             // 0..3
    const int lm    = lane & 15;
    const int khalf = (lane >> 4) << 3;     // 0 or 8
    const int qb    = blockIdx.x * 64;
    const int bh    = blockIdx.y;
    const int b     = bh >> 4;
    const int h     = bh & 15;

    {   // stage Q tile asynchronously (consumed after first wait+barrier)
        int r = tid >> 1;
        int c = (tid & 1) << 5;
        const u16* g = qkv + (size_t)(b * S + qb + r) * TD + h * DH + c;
#pragma unroll
        for (int q4 = 0; q4 < 4; q4++)
            async_load_b128(lds_off(&qs[r * 72 + c + q4 * 8]), g + q4 * 8);
    }

    f32x8 o[4];
#pragma unroll
    for (int d = 0; d < 4; d++) o[d] = zero8();
    float mrow[8], lrow[8];
#pragma unroll
    for (int e = 0; e < 8; e++) { mrow[e] = -3.0e38f; lrow[e] = 0.f; }

    const int mh   = (lane >> 4) << 3;
    const int kend = qb + 64;                   // causal extent, uniform per block
    for (int kb = 0; kb < kend; kb += 32) {
        __syncthreads();                        // prior reads of ks/vts done
        {   // stage K tile (async) + V tile (transposed through VGPRs)
            int r = tid >> 2;                   // key 0..31
            int c = (tid & 3) << 4;             // dh chunk
            const u16* gk = qkv + (size_t)(b * S + kb + r) * TD + D + h * DH + c;
            async_load_b128(lds_off(&ks[r * 72 + c]),     gk);
            async_load_b128(lds_off(&ks[r * 72 + c + 8]), gk + 8);
            const u16* gv = qkv + (size_t)(b * S + kb + r) * TD + 2 * D + h * DH + c;
            union { u32x4 q[2]; u16 hh[16]; } t;
            t.q[0] = *(const u32x4*)(gv);
            t.q[1] = *(const u32x4*)(gv + 8);
#pragma unroll
            for (int j = 0; j < 16; j++)
                vts[(c + j) * LDS_STRIDE + r] = t.hh[j];
        }
        wait_async();                           // Q (first iter) + K async done
        __syncthreads();

        // S = Q @ K^T  (16 q rows x 32 keys), contraction over DH=64 in 2 steps
        f32x8 s0 = zero8(), s1 = zero8();
#pragma unroll
        for (int kk = 0; kk < 64; kk += 32) {
            bf16x16 aq = ld_frag(&qs[(wave * 16 + lm) * 72 + kk + khalf]);
            bf16x16 b0 = ld_frag(&ks[(lm) * 72 + kk + khalf]);
            bf16x16 b1 = ld_frag(&ks[(16 + lm) * 72 + kk + khalf]);
            s0 = wmma_bf16(aq, b0, s0);
            s1 = wmma_bf16(aq, b1, s1);
        }

        // online softmax update (rows split across VGPR index; cols across 16 lanes)
#pragma unroll
        for (int e = 0; e < 8; e++) {
            int qg = qb + wave * 16 + mh + e;
            float v0 = ((kb + lm)      <= qg) ? s0[e] * 0.125f : -10000.0f;
            float v1 = ((kb + 16 + lm) <= qg) ? s1[e] * 0.125f : -10000.0f;
            float rm = fmaxf(v0, v1);
            rm = fmaxf(rm, __shfl_xor(rm, 1, 32));
            rm = fmaxf(rm, __shfl_xor(rm, 2, 32));
            rm = fmaxf(rm, __shfl_xor(rm, 4, 32));
            rm = fmaxf(rm, __shfl_xor(rm, 8, 32));
            float mn  = fmaxf(mrow[e], rm);
            float fac = __expf(mrow[e] - mn);
            float p0  = __expf(v0 - mn);
            float p1  = __expf(v1 - mn);
            float rs  = p0 + p1;
            rs += __shfl_xor(rs, 1, 32);
            rs += __shfl_xor(rs, 2, 32);
            rs += __shfl_xor(rs, 4, 32);
            rs += __shfl_xor(rs, 8, 32);
            lrow[e] = lrow[e] * fac + rs;
            mrow[e] = mn;
#pragma unroll
            for (int d = 0; d < 4; d++) o[d][e] *= fac;
            // transpose P (C layout -> A layout) through per-wave LDS region
            int m = mh + e;
            ps[(wave * 16 + m) * LDS_STRIDE + lm]      = f2bf(p0);
            ps[(wave * 16 + m) * LDS_STRIDE + 16 + lm] = f2bf(p1);
        }

        // O += P @ V  (same-wave LDS RAW: DS pipe is in-order per wave)
        bf16x16 ap = ld_frag(&ps[(wave * 16 + lm) * LDS_STRIDE + khalf]);
#pragma unroll
        for (int d = 0; d < 4; d++) {
            bf16x16 bv = ld_frag(&vts[(d * 16 + lm) * LDS_STRIDE + khalf]);
            o[d] = wmma_bf16(ap, bv, o[d]);
        }
    }

    // normalize + store
#pragma unroll
    for (int d = 0; d < 4; d++)
#pragma unroll
        for (int e = 0; e < 8; e++) {
            int row = b * S + qb + wave * 16 + mh + e;
            int col = h * DH + d * 16 + lm;
            aout[(size_t)row * D + col] = f2bf(o[d][e] / lrow[e]);
        }
}

// ---------------------------------------------------------------------------
// Host-side launch sequence
// ---------------------------------------------------------------------------
extern "C" void kernel_launch(void* const* d_in, const int* in_sizes, int n_in,
                              void* d_out, int out_size, void* d_ws, size_t ws_size,
                              hipStream_t stream) {
    (void)in_sizes; (void)n_in; (void)out_size; (void)ws_size;
    const int B = 2, S = 2048, D = 1024, DI = 4096;
    const int M = B * S;

    const float* hs     = (const float*)d_in[0];
    const float* ln1_g  = (const float*)d_in[1];
    const float* ln1_b  = (const float*)d_in[2];
    const float* attn_w = (const float*)d_in[3];
    const float* attn_b = (const float*)d_in[4];
    const float* proj_w = (const float*)d_in[5];
    const float* proj_b = (const float*)d_in[6];
    const float* ln2_g  = (const float*)d_in[7];
    const float* ln2_b  = (const float*)d_in[8];
    const float* fc_w   = (const float*)d_in[9];
    const float* fc_b   = (const float*)d_in[10];
    const float* fc2_w  = (const float*)d_in[11];
    const float* fc2_b  = (const float*)d_in[12];
    float* out = (float*)d_out;

    // workspace carve-up (bf16 buffers + one f32 buffer), 256B aligned
    char* base = (char*)d_ws;
    size_t off = 0;
    auto alloc_h = [&](size_t n) -> u16* {
        u16* p = (u16*)(base + off);
        off += ((n * sizeof(u16) + 255) / 256) * 256;
        return p;
    };
    u16* wqkv  = alloc_h((size_t)D * 3 * D);
    u16* wproj = alloc_h((size_t)D * D);
    u16* wfc   = alloc_h((size_t)D * DI);
    u16* wfc2  = alloc_h((size_t)DI * D);
    u16* xb    = alloc_h((size_t)M * D);    // ln1 out; reused for ln2 out
    u16* big   = alloc_h((size_t)M * DI);   // qkv (M*3D) then fc1 out (M*DI)
    u16* ab    = alloc_h((size_t)M * D);    // attention output
    float* hid = (float*)(base + off);      // f32 residual stream after attention
    off += (size_t)M * D * sizeof(float);

    // 0) weight conversion f32 -> bf16
    convert_kernel<<<dim3(2048), 256, 0, stream>>>(attn_w, wqkv, D * 3 * D);
    convert_kernel<<<dim3(1024), 256, 0, stream>>>(proj_w, wproj, D * D);
    convert_kernel<<<dim3(2048), 256, 0, stream>>>(fc_w,   wfc,   D * DI);
    convert_kernel<<<dim3(2048), 256, 0, stream>>>(fc2_w,  wfc2,  DI * D);

    // 1) x = LN1(hidden) -> bf16
    ln_kernel<<<dim3(M), 256, 0, stream>>>(hs, ln1_g, ln1_b, xb, D);

    // 2) qkv = x @ attn_w + attn_b -> bf16 [M, 3D]
    u16* qkvb = big;
    gemm_bf16<<<dim3(3 * D / 128, M / 128), 256, 0, stream>>>(
        xb, wqkv, attn_b, nullptr, nullptr, qkvb, M, 3 * D, D, 0);

    // 3) a = causal flash attention -> bf16 [M, D]
    attn_kernel<<<dim3(S / 64, B * 16), 128, 0, stream>>>(qkvb, ab);

    // 4) hid = hidden + a @ proj_w + proj_b -> f32
    gemm_bf16<<<dim3(D / 128, M / 128), 256, 0, stream>>>(
        ab, wproj, proj_b, hs, hid, nullptr, M, D, D, 2);

    // 5) m = LN2(hid) -> bf16
    u16* mb = xb;
    ln_kernel<<<dim3(M), 256, 0, stream>>>(hid, ln2_g, ln2_b, mb, D);

    // 6) h = gelu(m @ fc_w + fc_b) -> bf16 [M, DI]
    u16* hb = big;
    gemm_bf16<<<dim3(DI / 128, M / 128), 256, 0, stream>>>(
        mb, wfc, fc_b, nullptr, nullptr, hb, M, DI, D, 1);

    // 7) out = hid + h @ fc2_w + fc2_b -> f32
    gemm_bf16<<<dim3(D / 128, M / 128), 256, 0, stream>>>(
        hb, wfc2, fc2_b, hid, out, nullptr, M, D, DI, 2);
}
